// _TernaryExpert_36283883716716
// MI455X (gfx1250) — matrix operand order repaired and checked by reading, not compiled, not run on hardware
//
#include <hip/hip_runtime.h>
#include <hip/hip_bf16.h>

typedef _Float16 half8  __attribute__((ext_vector_type(8)));
typedef _Float16 half16 __attribute__((ext_vector_type(16)));
typedef float    float8 __attribute__((ext_vector_type(8)));
typedef unsigned int uint4v __attribute__((ext_vector_type(4)));
typedef int      int8v  __attribute__((ext_vector_type(8)));
typedef int      int4v  __attribute__((ext_vector_type(4)));

#define D_DIM 2048
#define I_DIM 5632
#define N_TOK 8192

#if __has_builtin(__builtin_amdgcn_tensor_load_to_lds)
#define HAVE_TDM 1
#else
#define HAVE_TDM 0
#endif

// ---- Tensor Data Mover: load a [tile_d1 x tile_d0] f16 tile into LDS with
// row padding 64->72 halves (pad_interval=4: every 32 DWORDs; pad_amount=3: 4 DWORDs).
// Descriptor bitfields per CDNA5 ISA §8.3 (group0) / §8.4 (group1).
#if HAVE_TDM
__device__ __forceinline__ void tdm_load_tile_f16(const _Float16* gptr, unsigned lds_off,
                                                  unsigned tensor_d0, unsigned tensor_d1,
                                                  unsigned tile_d0, unsigned tile_d1,
                                                  unsigned stride0) {
    unsigned long long ga = (unsigned long long)(size_t)gptr;
    uint4v g0;
    g0[0] = 1u;                                        // count=1, user descriptor
    g0[1] = lds_off;                                   // lds_addr (bytes)
    g0[2] = (unsigned)(ga & 0xffffffffu);              // global_addr[31:0]
    g0[3] = (unsigned)((ga >> 32) & 0x01ffffffu)       // global_addr[56:32]
            | (2u << 30);                              // type=2 ("image")
    int8v g1;
    g1[0] = (1 << 16)                                  // data_size=1 (2 bytes)
          | (1 << 20)                                  // pad_enable
          | (4 << 22)                                  // pad_interval: 32 DWORDs
          | (3 << 25);                                 // pad_amount: 4 DWORDs (8 halves)
    g1[1] = (int)((tensor_d0 & 0xffffu) << 16);        // tensor_dim0[15:0] @bit48
    g1[2] = (int)((tensor_d0 >> 16) | ((tensor_d1 & 0xffffu) << 16)); // dim0 hi | dim1 lo
    g1[3] = (int)((tensor_d1 >> 16) | (tile_d0 << 16));               // dim1 hi | tile_dim0
    g1[4] = (int)(tile_d1);                            // tile_dim1 | tile_dim2=0 (2D)
    g1[5] = (int)stride0;                              // tensor_dim0_stride[31:0]
    g1[6] = 0;                                         // stride0 hi | stride1 lo
    g1[7] = 0;                                         // stride1 hi
    int4v z4 = {0, 0, 0, 0};                           // groups 2/3 unused (2D tile)
    int8v z8 = {0, 0, 0, 0, 0, 0, 0, 0};
    // 6-arg builtin on this toolchain: (g0, g1, g2, g3, extra, cpol)
    __builtin_amdgcn_tensor_load_to_lds(g0, g1, z4, z4, z8, 0);
}
#endif

// ---- WMMA fragment loaders (LDS rows are K-contiguous, stride 72 halves) ----
__device__ __forceinline__ half16 load_frag_a(const _Float16* row_base, int ks, int lane) {
    const _Float16* p = row_base + ks * 32 + ((lane >> 4) << 3);
    half8 lo = *(const half8*)(p);
    half8 hi = *(const half8*)(p + 16);
    return __builtin_shufflevector(lo, hi, 0,1,2,3,4,5,6,7,8,9,10,11,12,13,14,15);
}
__device__ __forceinline__ half16 load_frag_b(const _Float16* row_base, int ks, int lane) {
    const _Float16* p = row_base + ks * 32 + ((lane >> 4) << 3);
    half8 lo = *(const half8*)(p);
    half8 hi = *(const half8*)(p + 16);
    return __builtin_shufflevector(lo, hi, 0,1,2,3,4,5,6,7,8,9,10,11,12,13,14,15);
}
__device__ __forceinline__ float8 wmma_f16(half16 a, half16 b, float8 c) {
    return __builtin_amdgcn_wmma_f32_16x16x32_f16(false, a, false, b, (short)0, c, false, false);
}

// ---------------- Kernel 0: x fp32 -> f16 ----------------
__global__ __launch_bounds__(256)
void cvt_x_kernel(const float* __restrict__ x, _Float16* __restrict__ x16) {
    size_t i = ((size_t)blockIdx.x * 256 + threadIdx.x) * 8;
    float4 a = *(const float4*)(x + i);
    float4 b = *(const float4*)(x + i + 4);
    half8 h = { (_Float16)a.x, (_Float16)a.y, (_Float16)a.z, (_Float16)a.w,
                (_Float16)b.x, (_Float16)b.y, (_Float16)b.z, (_Float16)b.w };
    *(half8*)(x16 + i) = h;
}

// ---------------- Kernel 1: fused gate/up + SiLU epilogue ----------------
// Block: 128(M) x 64(N of I), BK=64. 8 waves (4x2), each 32x32 for gate AND up.
__global__ __launch_bounds__(256)
void gateup_kernel(const _Float16* __restrict__ x16,
                   const signed char* __restrict__ gw,
                   const signed char* __restrict__ uw,
                   const float* __restrict__ gs_p,
                   const float* __restrict__ us_p,
                   _Float16* __restrict__ h16) {
    __shared__ _Float16 sA [128 * 72];
    __shared__ _Float16 sBg[ 64 * 72];
    __shared__ _Float16 sBu[ 64 * 72];

    const int t    = threadIdx.x;
    const int lane = t & 31;
    const int w    = t >> 5;
    const int wm   = w & 3;
    const int wn   = w >> 2;
    const int m0   = blockIdx.y * 128;
    const int n0   = blockIdx.x * 64;
    const float gs = *gs_p;
    const float us = *us_p;

    float8 accg[2][2] = {};
    float8 accu[2][2] = {};

    const int arow = t >> 3, acol = t & 7;
    const int brow = t >> 2, bcol = t & 3;
#if HAVE_TDM
    const unsigned sA_off = (unsigned)(size_t)(void*)&sA[0];
#endif

    for (int kb = 0; kb < D_DIM; kb += 64) {
        // ---- stage A (f16 copy): TDM if available, else manual ----
#if HAVE_TDM
        if (w == 0)
            tdm_load_tile_f16(x16 + (size_t)m0 * D_DIM + kb, sA_off,
                              D_DIM, N_TOK, 64, 128, D_DIM);
#else
        #pragma unroll
        for (int i = 0; i < 4; ++i) {
            int r = arow + 32 * i;
            *(half8*)(&sA[r * 72 + acol * 8]) =
                *(const half8*)(x16 + (size_t)(m0 + r) * D_DIM + kb + acol * 8);
        }
#endif
        // ---- stage B: int8 -> f16 on the fly, gate + up [64 x 64] ----
        {
            const size_t goff = (size_t)(n0 + brow) * D_DIM + kb + bcol * 16;
            union { int4 v; signed char b[16]; } pg, pu;
            pg.v = *(const int4*)(gw + goff);
            pu.v = *(const int4*)(uw + goff);
            _Float16* dg = &sBg[brow * 72 + bcol * 16];
            _Float16* du = &sBu[brow * 72 + bcol * 16];
            #pragma unroll
            for (int j = 0; j < 16; ++j) {
                dg[j] = (_Float16)pg.b[j];
                du[j] = (_Float16)pu.b[j];
            }
            if (kb + 64 < D_DIM) {
                __builtin_prefetch(gw + goff + 64, 0, 1);
                __builtin_prefetch(uw + goff + 64, 0, 1);
            }
        }
#if HAVE_TDM
        if (w == 0) __builtin_amdgcn_s_wait_tensorcnt(0);
#endif
        __syncthreads();

        #pragma unroll
        for (int ks = 0; ks < 2; ++ks) {
            half16 af[2], bg[2], bu[2];
            #pragma unroll
            for (int tm = 0; tm < 2; ++tm)
                af[tm] = load_frag_a(&sA[(wm * 32 + tm * 16 + (lane & 15)) * 72], ks, lane);
            #pragma unroll
            for (int tn = 0; tn < 2; ++tn) {
                bg[tn] = load_frag_b(&sBg[(wn * 32 + tn * 16 + (lane & 15)) * 72], ks, lane);
                bu[tn] = load_frag_b(&sBu[(wn * 32 + tn * 16 + (lane & 15)) * 72], ks, lane);
            }
            #pragma unroll
            for (int tm = 0; tm < 2; ++tm)
                #pragma unroll
                for (int tn = 0; tn < 2; ++tn) {
                    accg[tm][tn] = wmma_f16(af[tm], bg[tn], accg[tm][tn]);
                    accu[tm][tn] = wmma_f16(af[tm], bu[tn], accu[tm][tn]);
                }
        }
        __syncthreads();
    }

    #pragma unroll
    for (int tm = 0; tm < 2; ++tm)
        #pragma unroll
        for (int tn = 0; tn < 2; ++tn)
            #pragma unroll
            for (int r = 0; r < 8; ++r) {
                int m = m0 + wm * 32 + tm * 16 + ((lane >> 4) << 3) + r;
                int n = n0 + wn * 32 + tn * 16 + (lane & 15);
                float g = gs * accg[tm][tn][r];
                float u = us * accu[tm][tn][r];
                float sig = 1.0f / (1.0f + __expf(-g));
                h16[(size_t)m * I_DIM + n] = (_Float16)(g * sig * u);
            }
}

// ---------------- Kernel 2: down projection ----------------
// Block: 128(M) x 128(N of D), BK=64 over K=I. 8 waves (4x2), each 32x64.
__global__ __launch_bounds__(256)
void down_kernel(const _Float16* __restrict__ h16,
                 const signed char* __restrict__ dw,
                 const float* __restrict__ ds_p,
                 float* __restrict__ out) {
    __shared__ _Float16 sA[128 * 72];
    __shared__ _Float16 sB[128 * 72];

    const int t    = threadIdx.x;
    const int lane = t & 31;
    const int w    = t >> 5;
    const int wm   = w >> 1;
    const int wn   = w & 1;
    const int m0   = blockIdx.y * 128;
    const int n0   = blockIdx.x * 128;
    const float dsc = *ds_p;

    float8 acc[2][4] = {};

    const int arow = t >> 3, acol = t & 7;
    const int brow = t >> 2, bcol = t & 3;
#if HAVE_TDM
    const unsigned sA_off = (unsigned)(size_t)(void*)&sA[0];
#endif

    for (int kb = 0; kb < I_DIM; kb += 64) {
#if HAVE_TDM
        if (w == 0)
            tdm_load_tile_f16(h16 + (size_t)m0 * I_DIM + kb, sA_off,
                              I_DIM, N_TOK, 64, 128, I_DIM);
#else
        #pragma unroll
        for (int i = 0; i < 4; ++i) {
            int r = arow + 32 * i;
            *(half8*)(&sA[r * 72 + acol * 8]) =
                *(const half8*)(h16 + (size_t)(m0 + r) * I_DIM + kb + acol * 8);
        }
#endif
        #pragma unroll
        for (int i = 0; i < 2; ++i) {
            int r = brow + 64 * i;
            const size_t goff = (size_t)(n0 + r) * I_DIM + kb + bcol * 16;
            union { int4 v; signed char b[16]; } p;
            p.v = *(const int4*)(dw + goff);
            _Float16* d = &sB[r * 72 + bcol * 16];
            #pragma unroll
            for (int j = 0; j < 16; ++j) d[j] = (_Float16)p.b[j];
            if (kb + 64 < I_DIM) __builtin_prefetch(dw + goff + 64, 0, 1);
        }
#if HAVE_TDM
        if (w == 0) __builtin_amdgcn_s_wait_tensorcnt(0);
#endif
        __syncthreads();

        #pragma unroll
        for (int ks = 0; ks < 2; ++ks) {
            half16 af[2], bf[4];
            #pragma unroll
            for (int tm = 0; tm < 2; ++tm)
                af[tm] = load_frag_a(&sA[(wm * 32 + tm * 16 + (lane & 15)) * 72], ks, lane);
            #pragma unroll
            for (int tn = 0; tn < 4; ++tn)
                bf[tn] = load_frag_b(&sB[(wn * 64 + tn * 16 + (lane & 15)) * 72], ks, lane);
            #pragma unroll
            for (int tm = 0; tm < 2; ++tm)
                #pragma unroll
                for (int tn = 0; tn < 4; ++tn)
                    acc[tm][tn] = wmma_f16(af[tm], bf[tn], acc[tm][tn]);
        }
        __syncthreads();
    }

    #pragma unroll
    for (int tm = 0; tm < 2; ++tm)
        #pragma unroll
        for (int tn = 0; tn < 4; ++tn)
            #pragma unroll
            for (int r = 0; r < 8; ++r) {
                int m = m0 + wm * 32 + tm * 16 + ((lane >> 4) << 3) + r;
                int n = n0 + wn * 64 + tn * 16 + (lane & 15);
                out[(size_t)m * D_DIM + n] = dsc * acc[tm][tn][r];
            }
}

extern "C" void kernel_launch(void* const* d_in, const int* in_sizes, int n_in,
                              void* d_out, int out_size, void* d_ws, size_t ws_size,
                              hipStream_t stream) {
    const float*       x   = (const float*)d_in[0];
    const signed char* gw  = (const signed char*)d_in[1];
    const float*       gs  = (const float*)d_in[2];
    const signed char* uw  = (const signed char*)d_in[3];
    const float*       us  = (const float*)d_in[4];
    const signed char* dw  = (const signed char*)d_in[5];
    const float*       dsc = (const float*)d_in[6];

    _Float16* x16 = (_Float16*)d_ws;                 // N*D halves (32 MB)
    _Float16* h16 = x16 + (size_t)N_TOK * D_DIM;     // N*I halves (92 MB)

    cvt_x_kernel<<<(N_TOK * D_DIM) / (256 * 8), 256, 0, stream>>>(x, x16);

    dim3 g1(I_DIM / 64, N_TOK / 128);
    gateup_kernel<<<g1, 256, 0, stream>>>(x16, gw, uw, gs, us, h16);

    dim3 g2(D_DIM / 128, N_TOK / 128);
    down_kernel<<<g2, 256, 0, stream>>>(h16, dw, dsc, (float*)d_out);
}
